// CrystalGraphConvNet_63496796504225
// MI455X (gfx1250) — compile-verified, matrix operand order checked by reading, not run on hardware
//
#include <hip/hip_runtime.h>
#include <hip/hip_bf16.h>

// ---- problem constants (match reference) ----
#define FDIM   64
#define ZDIM   192          // 2F + NBR
#define ZP     200          // padded LDS row stride (halves): keeps 16B alignment, staggers banks
#define NATOMS 50000
#define NEDGES 800000
#define TPOOL  4096
#define NCONV  3

typedef __attribute__((ext_vector_type(16))) __bf16        v16bf;
typedef __attribute__((ext_vector_type(8)))  float          v8f;
typedef __attribute__((ext_vector_type(8)))  unsigned short us8;
typedef __attribute__((ext_vector_type(4)))  unsigned short us4;

union BFrag { us8 h[2]; v16bf bf; };

__device__ __forceinline__ unsigned short f2bf(float f) {
  unsigned u = __float_as_uint(f);
  u += 0x7FFFu + ((u >> 16) & 1u);            // round-to-nearest-even
  return (unsigned short)(u >> 16);
}
__device__ __forceinline__ float bf2f(unsigned short h) {
  return __uint_as_float(((unsigned)h) << 16);
}
// monotone order-preserving uint encoding of float (for atomicMax); 0 == "empty segment"
__device__ __forceinline__ unsigned fenc(float f) {
  unsigned u = __float_as_uint(f);
  return (u & 0x80000000u) ? ~u : (u | 0x80000000u);
}
__device__ __forceinline__ float fdec(unsigned e) {
  if (e == 0u) return 0.0f;                   // no edges -> max treated as 0 (isfinite fixup)
  unsigned u = (e & 0x80000000u) ? (e ^ 0x80000000u) : ~e;
  return __uint_as_float(u);
}

// -------------------- setup kernels --------------------
__global__ void k_embed(const int* __restrict__ xt, const float* __restrict__ emb,
                        float* __restrict__ x, float* __restrict__ cnt) {
  size_t gid = (size_t)blockIdx.x * 256 + threadIdx.x;      // N*F threads
  int i = (int)(gid >> 6), f = (int)(gid & 63);
  x[gid] = emb[(size_t)xt[i] * FDIM + f];
  if (gid < NATOMS) cnt[gid] = 0.0f;
}

__global__ void k_count(const int* __restrict__ ei, float* __restrict__ cnt) {
  int e = blockIdx.x * 256 + threadIdx.x;
  if (e < NEDGES) atomicAdd(&cnt[ei[e]], 1.0f);
}

__global__ void k_reset(float* __restrict__ xn, float* __restrict__ nodesum,
                        unsigned* __restrict__ nodemax, float* __restrict__ stats) {
  size_t gid = (size_t)blockIdx.x * 256 + threadIdx.x;      // N*F threads
  xn[gid] = 0.0f;
  if (gid < NATOMS) { nodesum[gid] = 0.0f; nodemax[gid] = 0u; }
  if (gid < 512) stats[gid] = 0.0f;
}

// -------------------- fused edge GEMM (WMMA bf16, async-pipelined) --------------------
// Persistent blocks (128 thr = 4 waves), grid-stride over 64-edge tiles.
// Per tile: edge_attr streamed via GLOBAL_LOAD_ASYNC_TO_LDS_B128 (issued one tile
// ahead, overlapping the WMMA compute), z assembled as bf16 in LDS, then
// core = z@Wc.T + bc via v_wmma_f32_16x16x32_bf16, fused with bf16 core store,
// BN-stat partials, filter logit and segment-max of logits.
__global__ __launch_bounds__(128) void k_edge_gemm(
    const float* __restrict__ x, const int* __restrict__ ei,
    const float* __restrict__ ea, const float* __restrict__ Wc,
    const float* __restrict__ bc, const float* __restrict__ Wf,
    const float* __restrict__ bfv, unsigned short* __restrict__ core,
    float* __restrict__ filt, unsigned* __restrict__ nodemax,
    float* __restrict__ colsum, float* __restrict__ colsq)
{
  __shared__ unsigned short zl[64 * ZP];         // 25.6 KB bf16 z-tile
  __shared__ unsigned short wb[6 * 4 * 32 * 16]; // 24.6 KB Wc in WMMA-fragment order
  __shared__ float eastage[64 * FDIM];           // 16.4 KB async-DMA staging (fp32)
  __shared__ float wfl[ZDIM];
  __shared__ int  ii[64], jj[64];
  __shared__ float cs[FDIM], cq[FDIM];

  const int tid    = threadIdx.x;
  const int NT     = NEDGES / 64;                // 12500 tiles
  const int stride = gridDim.x;

  if (tid < 64) { cs[tid] = 0.0f; cq[tid] = 0.0f; }
  for (int k = tid; k < ZDIM; k += 128) wfl[k] = Wf[k];

  // Wc -> bf16 fragments (built ONCE per persistent block).
  // B element map (32x16, 16-bit): N = lane%16, K = 32*kb + h + 8*(h>=8) + 8*(lane>=16)
  for (int id = tid; id < 768; id += 128) {
    int kb = id >> 7, rem = id & 127, nb = rem >> 5, L = rem & 31;
    int n = nb * 16 + (L & 15);
    int koff = (L >= 16) ? 8 : 0;
    unsigned short* dst = wb + id * 16;
    #pragma unroll
    for (int h = 0; h < 16; ++h) {
      int K = kb * 32 + h + ((h >= 8) ? 8 : 0) + koff;
      dst[h] = f2bf(Wc[(size_t)n * ZDIM + K]);   // B[k][n] = Wc[n][k]
    }
  }

  // LDS byte offset of the staging buffer (generic-pointer low 32 bits == LDS offset)
  const unsigned stage_lds = (unsigned)(size_t)(void*)eastage;

  // issue async DMA of edge_attr for this block's first tile
  {
    const int eb0 = blockIdx.x * 64;
    const char* g = (const char*)(ea + (size_t)eb0 * FDIM) + (size_t)tid * 16;
    unsigned l = stage_lds + (unsigned)tid * 16;
    #pragma unroll
    for (int k = 0; k < 8; ++k)
      asm volatile("global_load_async_to_lds_b128 %0, %1, off"
                   :: "v"(l + (unsigned)(k * 2048)), "v"(g + (size_t)k * 2048) : "memory");
  }
  __syncthreads();

  const int wave = tid >> 5, lane = tid & 31;
  const int erow = wave * 16;
  const int m16  = lane & 15;
  const int hi   = (lane >= 16) ? 8 : 0;

  for (int tile = blockIdx.x; tile < NT; tile += stride) {
    const int eBase = tile * 64;
    if (tid < 64) { ii[tid] = ei[eBase + tid]; jj[tid] = ei[NEDGES + eBase + tid]; }
    if (tile + stride < NT)
      __builtin_prefetch(ei + (tile + stride) * 64 + tid, 0, 2);  // next tile's indices
    __syncthreads();

    // z part 1: gather x_i | x_j (k = 0..127) from L2-resident x, convert to bf16
    for (int c = tid; c < 64 * 32; c += 128) {
      int e = c >> 5, q = c & 31, k = q * 4;
      const float* src = (k < 64) ? x + (size_t)ii[e] * FDIM + k
                                  : x + (size_t)jj[e] * FDIM + (k - 64);
      float4 v = *(const float4*)src;
      us4 o; o.x = f2bf(v.x); o.y = f2bf(v.y); o.z = f2bf(v.z); o.w = f2bf(v.w);
      *(us4*)(zl + e * ZP + k) = o;
    }

    asm volatile("s_wait_asynccnt 0" ::: "memory");   // this wave's DMA done
    __syncthreads();                                  // all waves' DMA visible

    // z part 2: edge_attr from staged LDS (k = 128..191)
    for (int c = tid; c < 64 * 16; c += 128) {
      int e = c >> 4, q = c & 15, k = 128 + q * 4;
      float4 v = *(const float4*)(eastage + e * FDIM + q * 4);
      us4 o; o.x = f2bf(v.x); o.y = f2bf(v.y); o.z = f2bf(v.z); o.w = f2bf(v.w);
      *(us4*)(zl + e * ZP + k) = o;
    }
    __syncthreads();

    // stage consumed: issue NEXT tile's DMA now so it overlaps the WMMA compute
    if (tile + stride < NT) {
      const int ebn = (tile + stride) * 64;
      const char* g = (const char*)(ea + (size_t)ebn * FDIM) + (size_t)tid * 16;
      unsigned l = stage_lds + (unsigned)tid * 16;
      #pragma unroll
      for (int k = 0; k < 8; ++k)
        asm volatile("global_load_async_to_lds_b128 %0, %1, off"
                     :: "v"(l + (unsigned)(k * 2048)), "v"(g + (size_t)k * 2048) : "memory");
    }

    // A fragments (16x32 bf16): M = lane%16, K = 32*kb + {0..7,16..23} + 8*(lane>=16)
    v16bf afr[6];
    {
      const unsigned short* zr = zl + (erow + m16) * ZP;
      #pragma unroll
      for (int kb = 0; kb < 6; ++kb) {
        const us8* ap = (const us8*)(zr + kb * 32 + hi);
        BFrag u; u.h[0] = ap[0]; u.h[1] = ap[2];   // +32B = K+16
        afr[kb] = u.bf;
      }
    }

    #pragma unroll
    for (int nb = 0; nb < 4; ++nb) {
      v8f acc = {};
      #pragma unroll
      for (int kb = 0; kb < 6; ++kb) {
        const us8* bp = (const us8*)(wb + (((kb * 4 + nb) * 32 + lane) << 4));
        BFrag u; u.h[0] = bp[0]; u.h[1] = bp[1];
        acc = __builtin_amdgcn_wmma_f32_16x16x32_bf16(false, afr[kb], false, u.bf,
                                                      (short)0, acc, false, false);
      }
      // C layout: N = lane%16, M = r + 8*(lane>=16)
      const int col = nb * 16 + m16;
      const float bcn = bc[col];
      float s = 0.0f, sq = 0.0f;
      #pragma unroll
      for (int r = 0; r < 8; ++r) {
        float v = acc[r] + bcn;
        size_t e = (size_t)(eBase + erow + r + hi);
        core[e * FDIM + col] = f2bf(v);
        s += v; sq += v * v;
      }
      atomicAdd(&cs[col], s);
      atomicAdd(&cq[col], sq);
    }

    // filter logit: split 192-dot across lane halves, combine via wave32 shuffle
    {
      const unsigned short* zr = zl + (erow + m16) * ZP;
      const int k0 = (lane < 16) ? 0 : 96;
      float p = 0.0f;
      for (int k = 0; k < 96; ++k) p += bf2f(zr[k0 + k]) * wfl[k0 + k];
      p += __shfl_down(p, 16);
      if (lane < 16) {
        float fv = p + bfv[0];
        int e = eBase + erow + m16;
        filt[e] = fv;
        atomicMax(&nodemax[ii[erow + m16]], fenc(fv));
      }
    }
    __syncthreads();   // protect zl/ii/jj before next iteration rewrites them
  }

  // flush per-block BN-stat partials once
  if (tid < FDIM) { atomicAdd(&colsum[tid], cs[tid]); atomicAdd(&colsq[tid], cq[tid]); }
}

// -------------------- BN finalize (shared for edge/node stats) --------------------
__global__ void k_bn_finalize(const float* __restrict__ s, const float* __restrict__ sq,
                              float* __restrict__ mu, float* __restrict__ rsig, float count) {
  int f = threadIdx.x;
  float m = s[f] / count;
  float v = sq[f] / count - m * m;
  mu[f] = m;
  rsig[f] = rsqrtf(fmaxf(v, 0.0f) + 1e-5f);
}

// -------------------- segment softmax: exp + segment sum --------------------
__global__ void k_expsum(const float* __restrict__ filt, const unsigned* __restrict__ nodemax,
                         const int* __restrict__ ei, float* __restrict__ expv,
                         float* __restrict__ nodesum) {
  int e = blockIdx.x * 256 + threadIdx.x;
  int i = ei[e];
  float ev = __expf(filt[e] - fdec(nodemax[i]));
  expv[e] = ev;
  atomicAdd(&nodesum[i], ev);
}

// -------------------- alpha-weighted scatter aggregation --------------------
// One thread per (edge, 4-feature group): 8B bf16 loads, alpha computed once.
__global__ void k_aggregate(const unsigned short* __restrict__ core,
                            const float* __restrict__ expv, const float* __restrict__ nodesum,
                            const int* __restrict__ ei, const float* __restrict__ mu,
                            const float* __restrict__ rsig, const float* __restrict__ g1,
                            const float* __restrict__ b1, float* __restrict__ xn) {
  size_t gid = (size_t)blockIdx.x * 256 + threadIdx.x;     // E*16 threads
  size_t e = gid >> 4;
  int f0 = (int)(gid & 15) * 4;
  int i = ei[e];
  float a = expv[e] / (nodesum[i] + 1e-16f);
  us4 c4 = *(const us4*)(core + e * FDIM + f0);
  float* dst = xn + (size_t)i * FDIM + f0;
  #pragma unroll
  for (int t = 0; t < 4; ++t) {
    int f = f0 + t;
    float v = fmaxf((bf2f(c4[t]) - mu[f]) * rsig[f] * g1[f] + b1[f], 0.0f);
    atomicAdd(&dst[t], v * a);
  }
}

// -------------------- node-side BN stats over agg = xn/denom --------------------
__global__ void k_node_stats(const float* __restrict__ xn, const float* __restrict__ cnt,
                             float* __restrict__ colsum2, float* __restrict__ colsq2) {
  __shared__ float s1[FDIM], s2[FDIM];
  int tid = threadIdx.x;
  if (tid < FDIM) { s1[tid] = 0.0f; s2[tid] = 0.0f; }
  __syncthreads();
  size_t gid = (size_t)blockIdx.x * 256 + tid;             // N*F threads
  int i = (int)(gid >> 6), f = tid & 63;
  float a = xn[gid] / fmaxf(cnt[i], 1.0f);
  atomicAdd(&s1[f], a);
  atomicAdd(&s2[f], a * a);
  __syncthreads();
  if (tid < FDIM) { atomicAdd(&colsum2[tid], s1[tid]); atomicAdd(&colsq2[tid], s2[tid]); }
}

// -------------------- x = relu(bn(agg) + x) --------------------
__global__ void k_update_x(const float* __restrict__ xn, const float* __restrict__ cnt,
                           const float* __restrict__ mu2, const float* __restrict__ rsig2,
                           const float* __restrict__ g2, const float* __restrict__ b2,
                           float* __restrict__ x) {
  size_t gid = (size_t)blockIdx.x * 256 + threadIdx.x;     // N*F threads
  int i = (int)(gid >> 6), f = threadIdx.x & 63;
  float a = xn[gid] / fmaxf(cnt[i], 1.0f);
  float v = (a - mu2[f]) * rsig2[f] * g2[f] + b2[f] + x[gid];
  x[gid] = fmaxf(v, 0.0f);
}

// -------------------- pooled head --------------------
__global__ void k_head(const float* __restrict__ x, const int* __restrict__ target,
                       const float* __restrict__ Wfc, const float* __restrict__ bfc,
                       const float* __restrict__ Ws, const float* __restrict__ bs,
                       float* __restrict__ out) {
  int t = blockIdx.x * 64 + threadIdx.x;
  if (t >= TPOOL) return;
  const float* xr = x + (size_t)target[t] * FDIM;
  float h[FDIM];
  #pragma unroll
  for (int k = 0; k < FDIM; ++k) h[k] = fmaxf(xr[k], 0.0f);
  float l0 = bs[0], l1 = bs[1];
  for (int o = 0; o < FDIM; ++o) {
    float a = bfc[o];
    const float* wr = Wfc + (size_t)o * FDIM;
    #pragma unroll
    for (int k = 0; k < FDIM; ++k) a += h[k] * wr[k];
    a = fmaxf(a, 0.0f);
    l0 += a * Ws[o];
    l1 += a * Ws[FDIM + o];
  }
  float mx = fmaxf(l0, l1);
  float e0 = __expf(l0 - mx), e1 = __expf(l1 - mx);
  float inv = 1.0f / (e0 + e1);
  out[t * 2 + 0] = e0 * inv;
  out[t * 2 + 1] = e1 * inv;
}

// -------------------- host orchestration --------------------
extern "C" void kernel_launch(void* const* d_in, const int* in_sizes, int n_in,
                              void* d_out, int out_size, void* d_ws, size_t ws_size,
                              hipStream_t stream) {
  const int*   x_types = (const int*)  d_in[0];
  const int*   eidx    = (const int*)  d_in[1];
  const float* eattr   = (const float*)d_in[2];
  const int*   target  = (const int*)  d_in[3];
  const float* emb     = (const float*)d_in[4];
  const float* Wc      = (const float*)d_in[5];
  const float* bc      = (const float*)d_in[6];
  const float* Wf      = (const float*)d_in[7];
  const float* bfv     = (const float*)d_in[8];
  const float* g1      = (const float*)d_in[9];
  const float* b1      = (const float*)d_in[10];
  const float* g2      = (const float*)d_in[11];
  const float* b2      = (const float*)d_in[12];
  const float* Wfc     = (const float*)d_in[13];
  const float* bfc     = (const float*)d_in[14];
  const float* Ws      = (const float*)d_in[15];
  const float* bs      = (const float*)d_in[16];
  float* out = (float*)d_out;

  // workspace carve (256B aligned)
  char* p = (char*)d_ws;
  auto carve = [&](size_t bytes) { void* r = (void*)p; p += (bytes + 255) & ~(size_t)255; return r; };
  float*          x        = (float*)         carve((size_t)NATOMS * FDIM * 4);
  float*          xn       = (float*)         carve((size_t)NATOMS * FDIM * 4);
  float*          cnt      = (float*)         carve((size_t)NATOMS * 4);
  unsigned short* core     = (unsigned short*)carve((size_t)NEDGES * FDIM * 2);   // bf16
  float*          filt     = (float*)         carve((size_t)NEDGES * 4);
  float*          expv     = (float*)         carve((size_t)NEDGES * 4);
  unsigned*       nodemax  = (unsigned*)      carve((size_t)NATOMS * 4);
  float*          nodesum  = (float*)         carve((size_t)NATOMS * 4);
  float*          stats    = (float*)         carve(512 * 4);
  float* colsum  = stats,       * colsq  = stats + 64,  * mu   = stats + 128, * rsig  = stats + 192;
  float* colsum2 = stats + 256, * colsq2 = stats + 320, * mu2  = stats + 384, * rsig2 = stats + 448;

  const int NF_BLOCKS   = (NATOMS * FDIM) / 256;                 // 12500
  const int E_BLOCKS    = NEDGES / 256;                          // 3125
  const int EF16_BLOCKS = (int)(((size_t)NEDGES * 16) / 256);    // 50000
  const int GEMM_BLOCKS = 2500;                                  // persistent: 5 tiles each

  k_embed<<<NF_BLOCKS, 256, 0, stream>>>(x_types, emb, x, cnt);
  k_count<<<E_BLOCKS, 256, 0, stream>>>(eidx, cnt);

  for (int l = 0; l < NCONV; ++l) {
    k_reset<<<NF_BLOCKS, 256, 0, stream>>>(xn, nodesum, nodemax, stats);
    k_edge_gemm<<<GEMM_BLOCKS, 128, 0, stream>>>(
        x, eidx, eattr,
        Wc + (size_t)l * FDIM * ZDIM, bc + (size_t)l * FDIM,
        Wf + (size_t)l * ZDIM, bfv + l,
        core, filt, nodemax, colsum, colsq);
    k_bn_finalize<<<1, 64, 0, stream>>>(colsum, colsq, mu, rsig, (float)NEDGES);
    k_expsum<<<E_BLOCKS, 256, 0, stream>>>(filt, nodemax, eidx, expv, nodesum);
    k_aggregate<<<EF16_BLOCKS, 256, 0, stream>>>(core, expv, nodesum, eidx, mu, rsig,
                                                 g1 + (size_t)l * FDIM, b1 + (size_t)l * FDIM, xn);
    k_node_stats<<<NF_BLOCKS, 256, 0, stream>>>(xn, cnt, colsum2, colsq2);
    k_bn_finalize<<<1, 64, 0, stream>>>(colsum2, colsq2, mu2, rsig2, (float)NATOMS);
    k_update_x<<<NF_BLOCKS, 256, 0, stream>>>(xn, cnt, mu2, rsig2,
                                              g2 + (size_t)l * FDIM, b2 + (size_t)l * FDIM, x);
  }
  k_head<<<TPOOL / 64, 64, 0, stream>>>(x, target, Wfc, bfc, Ws, bs, out);
}
